// QKVAttention_13709535609503
// MI455X (gfx1250) — matrix operand order repaired and checked by reading, not compiled, not run on hardware
//
#include <hip/hip_runtime.h>
#include <hip/hip_bf16.h>
#include <stdint.h>

typedef __attribute__((ext_vector_type(16))) __bf16 v16bf;
typedef __attribute__((ext_vector_type(8)))  float  v8f;

namespace {
constexpr int kNH = 8;      // heads
constexpr int kB  = 32;     // bs * heads
constexpr int kC  = 64;     // channels per head
constexpr int kL  = 1024;   // sequence length (T == S)
constexpr float kScale = 0.35355339059327373f;  // 64^(-1/4)
}

union BF16Frag { v16bf v; __bf16 e[16]; };
union F32Acc  { v8f   v; float  e[8];  };

__device__ __forceinline__ v8f wmma_bf16(v16bf a, v16bf b, v8f c) {
  // D = A(16x32 bf16) * B(32x16 bf16) + C(16x16 f32)
  return __builtin_amdgcn_wmma_f32_16x16x32_bf16(
      /*neg_a=*/false, a, /*neg_b=*/false, b,
      /*c_mod=*/(short)0, c, /*reuse_a=*/false, /*reuse_b=*/false);
}

// CDNA5 async global->LDS DMA path (guarded: falls back to plain loads).
// ROCm clang-22 signature (from probe): (int4 AS1*, int4 AS3*, imm, imm).
#if __has_builtin(__builtin_amdgcn_global_load_async_to_lds_b128) && \
    __has_builtin(__builtin_amdgcn_s_wait_asynccnt)
#define KQV_ASYNC_LDS 1
typedef int v4i_b __attribute__((vector_size(16)));
typedef __attribute__((address_space(1))) v4i_b* gptr16;
typedef __attribute__((address_space(3))) v4i_b* lptr16;
__device__ __forceinline__ void async_copy16(const void* g, void* lds) {
  __builtin_amdgcn_global_load_async_to_lds_b128(
      (gptr16)(uintptr_t)g, (lptr16)(uintptr_t)lds,
      /*offset=*/0, /*cpol=*/0);
}
#else
#define KQV_ASYNC_LDS 0
#endif

// ---------------------------------------------------------------------------
// Kernel 1: fp32 qkv -> bf16 Q[b][c][l] (scaled), Kt[b][s][c] (scaled,
// transposed so qk_scores B-fragments are lane-contiguous), V[b][c][s].
// ---------------------------------------------------------------------------
__global__ __launch_bounds__(256) void prep_qkv(
    const float* __restrict__ qkv,
    __bf16* __restrict__ Q, __bf16* __restrict__ Kt, __bf16* __restrict__ V) {
  size_t idx = (size_t)blockIdx.x * 256 + threadIdx.x;  // over kB*kC*kL
  int b   = (int)(idx / (kC * kL));
  int rem = (int)(idx % (kC * kL));          // c*kL + t
  int c = rem >> 10, t = rem & (kL - 1);
  int bsi = b / kNH, h = b % kNH;
  size_t src = ((size_t)bsi * 3 * kNH * kC + (size_t)h * kC) * kL + rem;
  size_t hc  = (size_t)kNH * kC * kL;        // one q/k/v slab per batch
  Q[idx] = (__bf16)(qkv[src] * kScale);
  Kt[((size_t)b * kL + t) * kC + c] = (__bf16)(qkv[src + hc] * kScale);
  V[idx] = (__bf16)(qkv[src + 2 * hc]);
}

// ---------------------------------------------------------------------------
// Kernel 2: S[b,t,s] = sum_c Q[b,c,t] * K[b,c,s]   (16-row t-tile per block)
// B-fragments from Kt are 32B-contiguous per lane -> global_load_b128.
// ---------------------------------------------------------------------------
__global__ __launch_bounds__(256) void qk_scores(
    const __bf16* __restrict__ Q, const __bf16* __restrict__ Kt,
    float* __restrict__ S) {
  const int b  = blockIdx.y;
  const int t0 = blockIdx.x * 16;
  __shared__ __bf16 qs[16][kC + 8];     // A tile: [t][c]
  for (int i = threadIdx.x; i < 16 * kC; i += 256) {
    int m = i >> 6, c = i & 63;
    qs[m][c] = Q[((size_t)b * kC + c) * kL + t0 + m];
  }
  __syncthreads();
  const int wave = threadIdx.x >> 5, lane = threadIdx.x & 31;
  const int g = lane >> 4, ln = lane & 15;
  BF16Frag a0, a1;                       // two K-steps over c
#pragma unroll
  for (int e = 0; e < 16; ++e) {
    int k = e + 16 * g;
    a0.e[e] = qs[ln][k];
    a1.e[e] = qs[ln][k + 32];
  }
  for (int st = wave; st < kL / 16; st += 8) {
    const int s0 = st * 16;
    const __bf16* kp = Kt + ((size_t)b * kL + s0 + ln) * kC + 16 * g;
    BF16Frag b0, b1;                     // B[k=c][n=s]
#pragma unroll
    for (int e = 0; e < 16; ++e) {
      b0.e[e] = kp[e];
      b1.e[e] = kp[e + 32];
    }
    v8f acc = {};
    acc = wmma_bf16(a0.v, b0.v, acc);
    acc = wmma_bf16(a1.v, b1.v, acc);
    F32Acc d; d.v = acc;
#pragma unroll
    for (int r = 0; r < 8; ++r) {
      int t = t0 + r + 8 * g;
      S[((size_t)b * kL + t) * kL + s0 + ln] = d.e[r];
    }
  }
}

// ---------------------------------------------------------------------------
// Kernel 3: R[s,b,t] = sum_c Q[b,c,s] * k_rp[s,t,c]  (GEMM at fixed s:
// (32b x 64c) @ (64c x 1024t); all 32 batches reuse the 256KB k_rp slab).
// Stores are coalesced in t; the transpose into [b,t,s] order happens in the
// fused softmax kernel through LDS (avoids 4B scattered RMW on 128MB).
// ---------------------------------------------------------------------------
__global__ __launch_bounds__(256) void rel_scores(
    const __bf16* __restrict__ Q, const float* __restrict__ k_rp,
    float* __restrict__ R) {
  const int s = blockIdx.x;
  __shared__ __bf16 qs[kB][kC + 8];     // A tile: [b][c] (Q column s)
  for (int i = threadIdx.x; i < kB * kC; i += 256) {
    int bb = i >> 6, c = i & 63;
    qs[bb][c] = Q[((size_t)bb * kC + c) * kL + s];
  }
  __syncthreads();
  const int wave = threadIdx.x >> 5, lane = threadIdx.x & 31;
  const int g = lane >> 4, ln = lane & 15;
  BF16Frag a[2][2];                      // [m-tile over b][k-step over c]
#pragma unroll
  for (int mi = 0; mi < 2; ++mi)
#pragma unroll
    for (int kk = 0; kk < 2; ++kk)
#pragma unroll
      for (int e = 0; e < 16; ++e) {
        int k = e + 16 * g + 32 * kk;
        a[mi][kk].e[e] = qs[mi * 16 + ln][k];
      }
  for (int tt = wave; tt < kL / 16; tt += 8) {
    const int t = tt * 16 + ln;          // B column n -> t
    const float* kp = k_rp + ((size_t)s * kL + t) * kC;  // contiguous in c
    if (tt + 8 < kL / 16)
      __builtin_prefetch(kp + (size_t)128 * kC, 0, 1);   // global_prefetch_b8
    BF16Frag b0, b1;
#pragma unroll
    for (int e = 0; e < 16; ++e) {
      int c = e + 16 * g;
      b0.e[e] = (__bf16)kp[c];
      b1.e[e] = (__bf16)kp[c + 32];
    }
#pragma unroll
    for (int mi = 0; mi < 2; ++mi) {
      v8f acc = {};
      acc = wmma_bf16(a[mi][0].v, b0.v, acc);
      acc = wmma_bf16(a[mi][1].v, b1.v, acc);
      F32Acc d; d.v = acc;
#pragma unroll
      for (int r = 0; r < 8; ++r) {
        int bb = mi * 16 + r + 8 * g;
        R[((size_t)s * kB + bb) * kL + tt * 16 + ln] = d.e[r];  // coalesced
      }
    }
  }
}

// ---------------------------------------------------------------------------
// Kernel 4: fused  w = S + R^T, mask, softmax -> bf16 P.
// One 1024-thread block owns a [32 t x 1024 s] fp32 stripe in dynamic LDS
// (~128KB; fits the 320KB WGP). R is staged with fully coalesced 128B reads
// and transposed through LDS; each wave then reduces its own t-row.
// ---------------------------------------------------------------------------
__global__ __launch_bounds__(1024) void rel_mask_softmax(
    const float* __restrict__ S, const float* __restrict__ R,
    const int* __restrict__ mask, __bf16* __restrict__ P) {
  extern __shared__ float wdyn[];        // [32][kL + 1]
  const int b  = blockIdx.y;
  const int t0 = blockIdx.x * 32;
  const int tid = threadIdx.x;
  // Stage R[s, b, t0..t0+31] -> wdyn[t][s]; lanes read consecutive t (128B).
  const int tt_ld = tid & 31, ss_ld = tid >> 5;
  for (int s0 = 0; s0 < kL; s0 += 32) {
    wdyn[(size_t)tt_ld * (kL + 1) + s0 + ss_ld] =
        R[((size_t)(s0 + ss_ld) * kB + b) * kL + t0 + tt_ld];
  }
  __syncthreads();
  const int wv = tid >> 5, lane = tid & 31;   // wave wv owns row t0+wv
  float* w = wdyn + (size_t)wv * (kL + 1);
  const size_t ro = ((size_t)b * kL + t0 + wv) * kL;
  float mx = -3.0e38f;
#pragma unroll 4
  for (int j = 0; j < kL / 32; ++j) {
    int s = lane + 32 * j;
    float val = (mask[ro + s] == 0) ? -1.0e8f : (w[s] + S[ro + s]);
    w[s] = val;
    mx = fmaxf(mx, val);
  }
#pragma unroll
  for (int off = 16; off > 0; off >>= 1)
    mx = fmaxf(mx, __shfl_xor(mx, off, 32));  // wave32 reduce
  float sum = 0.f;
#pragma unroll 4
  for (int j = 0; j < kL / 32; ++j) {
    int s = lane + 32 * j;
    float e = __expf(w[s] - mx);
    w[s] = e;
    sum += e;
  }
#pragma unroll
  for (int off = 16; off > 0; off >>= 1)
    sum += __shfl_xor(sum, off, 32);
  float inv = 1.0f / sum;
#pragma unroll 4
  for (int j = 0; j < kL / 32; ++j) {
    int s = lane + 32 * j;
    P[ro + s] = (__bf16)(w[s] * inv);
  }
}

// ---------------------------------------------------------------------------
// Kernel 5: out[b,c,t] = sum_s P[b,t,s] * V[b,c,s]  (K=1024 WMMA chain;
// both A and B fragment loads are lane-contiguous from global).
// ---------------------------------------------------------------------------
__global__ __launch_bounds__(128) void out_v(
    const __bf16* __restrict__ P, const __bf16* __restrict__ V,
    float* __restrict__ out) {
  const int b  = blockIdx.y;
  const int t0 = blockIdx.x * 16;
  const int wave = threadIdx.x >> 5;   // c-tile index (0..3)
  const int lane = threadIdx.x & 31;
  const int g = lane >> 4, ln = lane & 15;
  const int c = wave * 16 + ln;
  v8f acc = {};
  for (int s0 = 0; s0 < kL; s0 += 32) {
    BF16Frag a, bf;
    const __bf16* pp = P + ((size_t)b * kL + t0 + ln) * kL + s0;
    const __bf16* vp = V + ((size_t)b * kC + c) * kL + s0 + 16 * g;
#pragma unroll
    for (int e = 0; e < 16; ++e) {
      int ka = e + 16 * g;
      a.e[e]  = pp[ka];
      bf.e[e] = vp[e];
    }
    acc = wmma_bf16(a.v, bf.v, acc);
  }
  F32Acc d; d.v = acc;
#pragma unroll
  for (int r = 0; r < 8; ++r) {
    int t = t0 + r + 8 * g;
    out[((size_t)b * kC + c) * kL + t] = d.e[r];
  }
}

// ---------------------------------------------------------------------------
// Kernel 6: out[b,c,t] += sum_s P[b,t,s] * v_rp[t,s,c]  (GEMM at fixed t).
// Tiles are staged into LDS each k-step via CDNA5 async global->LDS b128
// DMA (ASYNCcnt + s_wait_asynccnt) when available; v_rp stays fp32 in LDS
// and is converted to bf16 at fragment read.
// ---------------------------------------------------------------------------
__global__ __launch_bounds__(256) void out_vrp(
    const __bf16* __restrict__ P, const float* __restrict__ v_rp,
    float* __restrict__ out) {
  const int t = blockIdx.x;
  __shared__ __bf16 pA[kB][40];         // [b][s-chunk], 80B rows (16B aligned)
  __shared__ float  vB[32][68];         // [s-chunk][c], 272B rows (16B aligned)
  const int tid = threadIdx.x;
  const int wave = tid >> 5, lane = tid & 31;
  const int mi = wave >> 2, ni = wave & 3;   // (b-tile, c-tile) per wave
  const int g = lane >> 4, ln = lane & 15;
  v8f acc = {};
  for (int s0 = 0; s0 < kL; s0 += 32) {
    __syncthreads();                     // protect prior-iteration reads
#if KQV_ASYNC_LDS
    if (tid < 128) {                     // pA: 32 rows x 4 chunks x 16B
      int bb = tid >> 2, ck = tid & 3;
      async_copy16(&P[((size_t)bb * kL + t) * kL + s0 + ck * 8], &pA[bb][ck * 8]);
    }
#pragma unroll
    for (int u = 0; u < 2; ++u) {        // vB: 32 rows x 16 chunks x 16B
      int i = tid * 2 + u;
      int ss = i >> 4, ck = i & 15;
      async_copy16(&v_rp[((size_t)t * kL + s0 + ss) * kC + ck * 4], &vB[ss][ck * 4]);
    }
    __builtin_amdgcn_s_wait_asynccnt(0);
    __syncthreads();
#else
    for (int i = tid; i < kB * 32; i += 256) {
      int bb = i >> 5, ss = i & 31;
      pA[bb][ss] = P[((size_t)bb * kL + t) * kL + s0 + ss];
    }
    for (int i = tid; i < 32 * kC; i += 256) {
      int ss = i >> 6, cc = i & 63;
      vB[ss][cc] = v_rp[((size_t)t * kL + s0 + ss) * kC + cc];
    }
    __syncthreads();
#endif
    BF16Frag a, bf;
#pragma unroll
    for (int e = 0; e < 16; ++e) {
      int ka = e + 16 * g;
      a.e[e]  = pA[mi * 16 + ln][ka];
      int kb = e + 16 * g;
      bf.e[e] = (__bf16)vB[kb][ni * 16 + ln];
    }
    acc = wmma_bf16(a.v, bf.v, acc);
  }
  F32Acc d; d.v = acc;
#pragma unroll
  for (int r = 0; r < 8; ++r) {
    int bb = mi * 16 + r + 8 * g;
    int c  = ni * 16 + ln;
    out[((size_t)bb * kC + c) * kL + t] += d.e[r];
  }
}

// ---------------------------------------------------------------------------
extern "C" void kernel_launch(void* const* d_in, const int* in_sizes, int n_in,
                              void* d_out, int out_size, void* d_ws, size_t ws_size,
                              hipStream_t stream) {
  const float* qkv  = (const float*)d_in[0];
  const float* k_rp = (const float*)d_in[1];
  const float* v_rp = (const float*)d_in[2];
  const int*   mask = (const int*)d_in[3];
  float* out = (float*)d_out;

  char* ws = (char*)d_ws;
  const size_t qkvBytes = (size_t)kB * kC * kL * 2;        // 4 MB each
  const size_t sBytes   = (size_t)kB * kL * kL * 4;        // 128 MB fp32
  __bf16* Qb  = (__bf16*)ws;
  __bf16* Ktb = (__bf16*)(ws + qkvBytes);
  __bf16* Vb  = (__bf16*)(ws + 2 * qkvBytes);
  float*  Sb  = (float*)(ws + 3 * qkvBytes);                // qk scores
  float*  Rb  = (float*)(ws + 3 * qkvBytes + sBytes);       // rel scores [s][b][t]
  __bf16* Pb  = (__bf16*)(ws + 3 * qkvBytes + 2 * sBytes);  // 64 MB probs

  const uint32_t smaxLds = 32 * (kL + 1) * sizeof(float);   // ~128 KB

  prep_qkv        <<<dim3((kB * kC * kL) / 256), dim3(256),  0,       stream>>>(qkv, Qb, Ktb, Vb);
  qk_scores       <<<dim3(kL / 16, kB),          dim3(256),  0,       stream>>>(Qb, Ktb, Sb);
  rel_scores      <<<dim3(kL),                   dim3(256),  0,       stream>>>(Qb, k_rp, Rb);
  rel_mask_softmax<<<dim3(kL / 32, kB),          dim3(1024), smaxLds, stream>>>(Sb, Rb, mask, Pb);
  out_v           <<<dim3(kL / 16, kB),          dim3(128),  0,       stream>>>(Pb, Vb, out);
  out_vrp         <<<dim3(kL),                   dim3(256),  0,       stream>>>(Pb, v_rp, out);
}
